// LocalRNN_37967510897054
// MI455X (gfx1250) — matrix operand order repaired and checked by reading, not compiled, not run on hardware
//
#include <hip/hip_runtime.h>
#include <stdint.h>

// ---------------- problem constants ----------------
#define BN 32
#define LN 2048
#define DN 512
#define GN 1536            // 3*DN (r,z,n gates)
#define TL 32              // output positions per workgroup
#define XR 34              // TL + KSIZE-1 rows of gi needed
#define AR 48              // A-buffer rows, padded to 3 m-tiles (rows >= XR zero-filled)
#define AP 520             // A row stride in bf16 elems (1040 B, 16B aligned, bank-skewed)
#define GP 1540            // gi row stride in f32 elems (6160 B)
#define NTHREADS 256       // 8 wave32s

typedef __attribute__((ext_vector_type(16))) __bf16 v16bf;
typedef __attribute__((ext_vector_type(8)))  __bf16 v8bf;
typedef __attribute__((ext_vector_type(4)))  __bf16 v4bf;
typedef __attribute__((ext_vector_type(8)))  float  v8f;

// ---------------- dynamic LDS layout (bytes) ----------------
#define OFF_A0   0
#define SZ_A0    (AR * AP * 2)            // 49,920  : x tile (bf16), later h2
#define OFF_A1   (OFF_A0 + SZ_A0)
#define SZ_A1    (TL * AP * 2)            // 33,280  : h1 (bf16)
#define OFF_GI   (OFF_A1 + SZ_A1)
#define SZ_GI    (XR * GP * 4)            // 209,440 : gi tile (f32); also TDM x-stage buffer
#define OFF_BIAS (OFF_GI + SZ_GI)
#define SZ_BIAS  (2 * GN * 4)             // 12,288  : b_ih | b_hh (f32)
#define SMEM_BYTES (OFF_BIAS + SZ_BIAS)   // 304,928 < 320 KB WGP LDS

#if defined(__AMDGCN__) && __has_builtin(__builtin_amdgcn_tensor_load_to_lds) && \
    __has_builtin(__builtin_amdgcn_s_wait_tensorcnt)
#define USE_TDM 1
#else
#define USE_TDM 0
#endif

__device__ __forceinline__ float sigmoidf_(float x) {
  return 1.0f / (1.0f + __expf(-x));
}

// A fragment: 16x32 bf16, row-major A in LDS (ISA 7.12.2 16-bit A layout).
__device__ __forceinline__ v16bf load_a_frag(const __bf16* A, int mbase, int kt, int lane) {
  const int row = mbase + (lane & 15);
  const int klo = (lane < 16) ? 0 : 8;
  const __bf16* p = A + row * AP + kt * 32 + klo;
  v8bf lo = *(const v8bf*)(p);          // K = klo .. klo+7
  v8bf hi = *(const v8bf*)(p + 16);     // K = klo+16 .. klo+23
  return __builtin_shufflevector(lo, hi, 0,1,2,3,4,5,6,7,8,9,10,11,12,13,14,15);
}

// B fragment: 32x16 bf16 where B[k][n] = W[grow+n][k] (row-major W == column-major B).
__device__ __forceinline__ v16bf load_b_frag(const __bf16* __restrict__ W, int grow, int kt, int lane) {
  const int g  = grow + (lane & 15);
  const int kk = kt * 32 + ((lane < 16) ? 0 : 8);
  const __bf16* p = W + (size_t)g * DN + kk;
  v8bf lo = *(const v8bf*)(p);
  v8bf hi = *(const v8bf*)(p + 16);
  return __builtin_shufflevector(lo, hi, 0,1,2,3,4,5,6,7,8,9,10,11,12,13,14,15);
}

__device__ __forceinline__ v8f wmma_bf16(v16bf a, v16bf b, v8f c) {
  return __builtin_amdgcn_wmma_f32_16x16x32_bf16(
      false, a, false, b, (short)0, c, false, false);
}

#if USE_TDM
typedef uint32_t u32x4 __attribute__((ext_vector_type(4)));
typedef int      i32x4 __attribute__((ext_vector_type(4)));
typedef int      i32x8 __attribute__((ext_vector_type(8)));

// TDM load of a 2D f32 tile [rows x width] (row stride == width) global -> LDS.
// D# bitfields per CDNA5 ISA 8.3/8.4.
__device__ __forceinline__ void tdm_load_2d_f32(const float* gsrc, uint32_t lds_off,
                                                int width, int rows) {
  const uint64_t ga = (uint64_t)(uintptr_t)gsrc;
  u32x4 g0;
  g0[0] = 1u;                                                  // count=1, user mode
  g0[1] = lds_off;                                             // lds_addr (bytes)
  g0[2] = (uint32_t)ga;                                        // global_addr[31:0]
  g0[3] = (uint32_t)((ga >> 32) & 0x1FFFFFFu) | (2u << 30);    // addr[56:32] | type=2
  i32x8 g1;
  g1[0] = 0x20000;                                             // data_size code 2 (4 bytes)
  g1[1] = (int)(((uint32_t)width & 0xFFFFu) << 16);            // tensor_dim0[15:0]
  g1[2] = (int)(((uint32_t)width >> 16) |
                (((uint32_t)rows & 0xFFFFu) << 16));           // dim0[31:16] | dim1[15:0]
  g1[3] = (int)(((uint32_t)rows >> 16) |
                (((uint32_t)width & 0xFFFFu) << 16));          // dim1[31:16] | tile_dim0
  g1[4] = (int)((uint32_t)rows & 0xFFFFu);                     // tile_dim1 (tile_dim2=0)
  g1[5] = width;                                               // tensor_dim0_stride[31:0]
  g1[6] = 0;
  g1[7] = 0;
  i32x4 z4 = {0, 0, 0, 0};
#if __clang_major__ >= 23
  i32x8 z8 = {0, 0, 0, 0, 0, 0, 0, 0};
  __builtin_amdgcn_tensor_load_to_lds(g0, g1, z4, z4, z8, 0);
#else
  __builtin_amdgcn_tensor_load_to_lds(g0, g1, z4, z4, 0);
#endif
}
#endif  // USE_TDM

// ---------------- phase 0: f32 -> bf16 weight conversion ----------------
__global__ __launch_bounds__(NTHREADS) void localrnn_wcvt(
    const float* __restrict__ wih, const float* __restrict__ whh,
    __bf16* __restrict__ wihb, __bf16* __restrict__ whhb) {
  int i = blockIdx.x * NTHREADS + threadIdx.x;
  if (i < GN * DN) {
    wihb[i] = (__bf16)wih[i];
    whhb[i] = (__bf16)whh[i];
  }
}

// ---------------- main fused LocalRNN kernel ----------------
// LDS already caps occupancy at 1 WG/WGP (2 waves/SIMD): give the allocator a big
// VGPR budget so current+next weight fragments can live simultaneously.
__global__ __launch_bounds__(NTHREADS, 1) __attribute__((amdgpu_waves_per_eu(1, 2)))
void localrnn_main(
    const float* __restrict__ x,
    const __bf16* __restrict__ Wih, const __bf16* __restrict__ Whh,
    const float* __restrict__ bih, const float* __restrict__ bhh,
    float* __restrict__ out) {
  extern __shared__ char smem[];
  __bf16* A0     = (__bf16*)(smem + OFF_A0);   // x tile (bf16), later h2
  __bf16* A1     = (__bf16*)(smem + OFF_A1);   // h1
  float*  gi     = (float*)(smem + OFF_GI);
  float*  xstage = (float*)(smem + OFF_GI);    // TDM f32 x-stage (dead before gi GEMM)
  float*  bias   = (float*)(smem + OFF_BIAS);  // [0,GN)=b_ih, [GN,2GN)=b_hh

  const int tid  = threadIdx.x;
  const int lane = tid & 31;
  const int wave = tid >> 5;
  const int lt   = blockIdx.x;   // 0..63 position tiles
  const int b    = blockIdx.y;   // 0..31 batch
  const int l0   = lt * TL;
  const int pad  = (l0 == 0) ? 2 : 0;     // left zero-pad rows (xp padding)

  const float* xb = x + (size_t)b * LN * DN;

#if USE_TDM
  // ---- Tensor Data Mover: DMA biases and the raw f32 x tile into LDS ----
  if (wave == 0) {
    tdm_load_2d_f32(bih, (uint32_t)(uintptr_t)bias, GN, 1);
    tdm_load_2d_f32(bhh, (uint32_t)(uintptr_t)(bias + GN), GN, 1);
    tdm_load_2d_f32(xb + (size_t)(l0 - 2 + pad) * DN,
                    (uint32_t)(uintptr_t)(xstage + pad * DN),
                    DN, XR - pad);
    __builtin_amdgcn_s_wait_tensorcnt(0);
  }
  __syncthreads();
  // convert staged f32 x -> bf16 A0, zero-filling pad rows and rows >= XR
  for (int i = tid; i < AR * (DN / 4); i += NTHREADS) {
    const int row = i / (DN / 4);
    const int c4  = (i % (DN / 4)) * 4;
    float4 v = make_float4(0.f, 0.f, 0.f, 0.f);
    if (row >= pad && row < XR) v = *(const float4*)(xstage + row * DN + c4);
    v4bf o;
    o[0] = (__bf16)v.x; o[1] = (__bf16)v.y; o[2] = (__bf16)v.z; o[3] = (__bf16)v.w;
    *(v4bf*)(A0 + row * AP + c4) = o;
  }
#else
  // ---- fallback: direct global staging ----
  for (int i = tid; i < GN; i += NTHREADS) {
    bias[i]      = bih[i];
    bias[GN + i] = bhh[i];
  }
  for (int i = tid; i < AR * (DN / 4); i += NTHREADS) {
    const int row = i / (DN / 4);
    const int c4  = (i % (DN / 4)) * 4;
    const int gl  = l0 - 2 + row;
    float4 v = make_float4(0.f, 0.f, 0.f, 0.f);
    if (row < XR && gl >= 0) v = *(const float4*)(xb + (size_t)gl * DN + c4);
    v4bf o;
    o[0] = (__bf16)v.x; o[1] = (__bf16)v.y; o[2] = (__bf16)v.z; o[3] = (__bf16)v.w;
    *(v4bf*)(A0 + row * AP + c4) = o;
  }
#endif
  __syncthreads();

  // ---- gi GEMM: one d-trio task owns all 3 m-tiles; 9 accumulators.
  //      Weight fragments double-buffered in registers: k+1's loads issue before
  //      k's WMMAs, so each iteration only waits on loads a full iteration old. ----
  for (int dt = wave; dt < 32; dt += 8) {
    const int dt16 = dt * 16;
    v8f aR0 = {}, aZ0 = {}, aN0 = {};
    v8f aR1 = {}, aZ1 = {}, aN1 = {};
    v8f aR2 = {}, aZ2 = {}, aN2 = {};
    v16bf bR = load_b_frag(Wih,          dt16, 0, lane);
    v16bf bZ = load_b_frag(Wih, DN     + dt16, 0, lane);
    v16bf bNn= load_b_frag(Wih, 2 * DN + dt16, 0, lane);
#pragma unroll 1
    for (int kt = 0; kt < 16; ++kt) {
      const int ktn = (kt < 15) ? kt + 1 : kt;   // branch-free tail
      v16bf nR = load_b_frag(Wih,          dt16, ktn, lane);
      v16bf nZ = load_b_frag(Wih, DN     + dt16, ktn, lane);
      v16bf nN = load_b_frag(Wih, 2 * DN + dt16, ktn, lane);
      v16bf f0 = load_a_frag(A0, 0,  kt, lane);
      v16bf f1 = load_a_frag(A0, 16, kt, lane);
      v16bf f2 = load_a_frag(A0, 32, kt, lane);
      aR0 = wmma_bf16(f0, bR,  aR0); aR1 = wmma_bf16(f1, bR,  aR1); aR2 = wmma_bf16(f2, bR,  aR2);
      aZ0 = wmma_bf16(f0, bZ,  aZ0); aZ1 = wmma_bf16(f1, bZ,  aZ1); aZ2 = wmma_bf16(f2, bZ,  aZ2);
      aN0 = wmma_bf16(f0, bNn, aN0); aN1 = wmma_bf16(f1, bNn, aN1); aN2 = wmma_bf16(f2, bNn, aN2);
      bR = nR; bZ = nZ; bNn = nN;
    }
    // D layout: lanes 0-15 hold M=0..7 (vgpr r), lanes 16-31 hold M=8..15
    const int n  = lane & 15;
    const int mo = (lane >> 4) * 8;
    const int c  = dt16 + n;
    const float bR_ = bias[c], bZ_ = bias[DN + c], bN_ = bias[2 * DN + c];
#pragma unroll
    for (int r = 0; r < 8; ++r) {
      const int row0 = mo + r;            // 0..15  : always < XR
      const int row1 = 16 + mo + r;       // 16..31 : always < XR
      const int row2 = 32 + mo + r;       // 32..47 : clamp to XR
      gi[row0 * GP + c]          = aR0[r] + bR_;
      gi[row0 * GP + DN + c]     = aZ0[r] + bZ_;
      gi[row0 * GP + 2*DN + c]   = aN0[r] + bN_;
      gi[row1 * GP + c]          = aR1[r] + bR_;
      gi[row1 * GP + DN + c]     = aZ1[r] + bZ_;
      gi[row1 * GP + 2*DN + c]   = aN1[r] + bN_;
      if (row2 < XR) {
        gi[row2 * GP + c]        = aR2[r] + bR_;
        gi[row2 * GP + DN + c]   = aZ2[r] + bZ_;
        gi[row2 * GP + 2*DN + c] = aN2[r] + bN_;
      }
    }
  }
  __syncthreads();

  // ---- t = 0: h == 0  =>  gh = b_hh; pure elementwise, h1 -> A1 ----
  for (int i = tid; i < TL * DN; i += NTHREADS) {
    const int j = i >> 9;
    const int d = i & (DN - 1);
    const float ir  = gi[j * GP + d];
    const float iz  = gi[j * GP + DN + d];
    const float in_ = gi[j * GP + 2*DN + d];
    const float rg = sigmoidf_(ir + bias[GN + d]);
    const float zg = sigmoidf_(iz + bias[GN + DN + d]);
    const float ng = tanhf(in_ + rg * bias[GN + 2*DN + d]);
    A1[j * AP + d] = (__bf16)((1.0f - zg) * ng);   // h1 = (1-z)*n + z*0
  }
  __syncthreads();

  // ---- t = 1,2: gh GEMM, one d-trio task owns both m-tiles; fused gate math ----
  float* outb = out + ((size_t)b * LN + l0) * DN;
  for (int t = 1; t <= 2; ++t) {
    __bf16* Ain  = (t == 1) ? A1 : A0;   // h_t (bf16)
    __bf16* Aout = A0;                   // h2 overwrites the x buffer at t==1
    for (int dt = wave; dt < 32; dt += 8) {
      const int dt16 = dt * 16;
      v8f hR0 = {}, hZ0 = {}, hN0 = {};
      v8f hR1 = {}, hZ1 = {}, hN1 = {};
      v16bf bR = load_b_frag(Whh,          dt16, 0, lane);
      v16bf bZ = load_b_frag(Whh, DN     + dt16, 0, lane);
      v16bf bNn= load_b_frag(Whh, 2 * DN + dt16, 0, lane);
#pragma unroll 1
      for (int kt = 0; kt < 16; ++kt) {
        const int ktn = (kt < 15) ? kt + 1 : kt;
        v16bf nR = load_b_frag(Whh,          dt16, ktn, lane);
        v16bf nZ = load_b_frag(Whh, DN     + dt16, ktn, lane);
        v16bf nN = load_b_frag(Whh, 2 * DN + dt16, ktn, lane);
        v16bf f0 = load_a_frag(Ain, 0,  kt, lane);
        v16bf f1 = load_a_frag(Ain, 16, kt, lane);
        hR0 = wmma_bf16(f0, bR,  hR0); hR1 = wmma_bf16(f1, bR,  hR1);
        hZ0 = wmma_bf16(f0, bZ,  hZ0); hZ1 = wmma_bf16(f1, bZ,  hZ1);
        hN0 = wmma_bf16(f0, bNn, hN0); hN1 = wmma_bf16(f1, bNn, hN1);
        bR = nR; bZ = nZ; bNn = nN;
      }
      const int n  = lane & 15;
      const int mo = (lane >> 4) * 8;
      const int d  = dt16 + n;
      const float bhr = bias[GN + d];
      const float bhz = bias[GN + DN + d];
      const float bhn = bias[GN + 2*DN + d];
#pragma unroll
      for (int mt = 0; mt < 2; ++mt) {
        const v8f cR = mt ? hR1 : hR0;
        const v8f cZ = mt ? hZ1 : hZ0;
        const v8f cN = mt ? hN1 : hN0;
#pragma unroll
        for (int r = 0; r < 8; ++r) {
          const int j = mt * 16 + mo + r;                // local position
          const float ir  = gi[(j + t) * GP + d];
          const float iz  = gi[(j + t) * GP + DN + d];
          const float in_ = gi[(j + t) * GP + 2*DN + d];
          const float hp  = (float)Ain[j * AP + d];      // h_t
          const float rg = sigmoidf_(ir + cR[r] + bhr);
          const float zg = sigmoidf_(iz + cZ[r] + bhz);
          const float ng = tanhf(in_ + rg * (cN[r] + bhn));
          const float hn = (1.0f - zg) * ng + zg * hp;
          if (t == 1) Aout[j * AP + d] = (__bf16)hn;     // h2
          else        outb[(size_t)j * DN + d] = hn;     // final h3 -> HBM
        }
      }
    }
    __syncthreads();
  }
}

// ---------------- host-side launcher ----------------
extern "C" void kernel_launch(void* const* d_in, const int* in_sizes, int n_in,
                              void* d_out, int out_size, void* d_ws, size_t ws_size,
                              hipStream_t stream) {
  const float* x   = (const float*)d_in[0];
  const float* wih = (const float*)d_in[1];
  const float* whh = (const float*)d_in[2];
  const float* bih = (const float*)d_in[3];
  const float* bhh = (const float*)d_in[4];
  float* out = (float*)d_out;

  // workspace: bf16 copies of W_ih and W_hh (3 MB total, L2-resident)
  __bf16* wihb = (__bf16*)d_ws;
  __bf16* whhb = wihb + (size_t)GN * DN;

  const int nW = GN * DN;  // 786,432 per matrix
  localrnn_wcvt<<<(nW + NTHREADS - 1) / NTHREADS, NTHREADS, 0, stream>>>(wih, whh, wihb, whhb);

  dim3 grid(LN / TL, BN);  // 64 x 32 workgroups
  localrnn_main<<<grid, NTHREADS, SMEM_BYTES, stream>>>(x, wihb, whhb, bih, bhh, out);
}
